// MyModel_61933428415780
// MI455X (gfx1250) — 8-bit matrix operand order repaired and checked by reading, compiled, not run on hardware
//
#include <hip/hip_runtime.h>
#include <hip/hip_bf16.h>

typedef __attribute__((ext_vector_type(16))) int   v16i;
typedef __attribute__((ext_vector_type(8)))  float v8f;
typedef __attribute__((ext_vector_type(4)))  int   v4i;

#define M_DIM 8192
#define N_DIM 8192
#define K_DIM 1024

#define BLK_M 128
#define BLK_N 128
#define BLK_K 128
#define PA 144
#define PB 144

typedef __attribute__((address_space(1))) v4i* gas_v4i_ptr;
typedef __attribute__((address_space(3))) v4i* lds_v4i_ptr;

#if __has_builtin(__builtin_amdgcn_global_load_async_to_lds_b128)
#define ASYNC_CP16(l, g)                                                      \
    __builtin_amdgcn_global_load_async_to_lds_b128(                           \
        (gas_v4i_ptr)(g), (lds_v4i_ptr)(l), 0, 0)
#else
#define ASYNC_CP16(l, g) (*(uint4*)(l) = *(const uint4*)(g))
#endif

#if __has_builtin(__builtin_amdgcn_s_wait_asynccnt)
#define WAIT_ASYNC() __builtin_amdgcn_s_wait_asynccnt(0)
#else
#define WAIT_ASYNC() asm volatile("s_wait_asynccnt 0x0" ::: "memory")
#endif

__device__ __forceinline__ unsigned int f32_to_e4m3(float x) {
    unsigned int u = __float_as_uint(x);
    unsigned int s = u >> 31;
    unsigned int a = u & 0x7FFFFFFFu;
    if (a >= 0x7F800000u) return (s << 7) | 0x7Fu;
    float af = __uint_as_float(a);
    if (af > 448.0f) return (s << 7) | 0x7Eu;
    unsigned int exp = a >> 23;
    if (exp >= 121u) {
        unsigned int r = a + 0x7FFFFu + ((a >> 20) & 1u);
        unsigned int e = (r >> 23) - 127u + 7u;
        unsigned int m = (r >> 20) & 7u;
        if (e >= 15u && m >= 7u) return (s << 7) | 0x7Eu;
        return (s << 7) | (e << 3) | m;
    }
    int m = (int)rintf(af * 512.0f);
    if (m > 7) return (s << 7) | 0x08u;
    return (s << 7) | (unsigned int)m;
}

__device__ __forceinline__ unsigned int pack4_e4m3(float a, float b, float c, float d) {
    return f32_to_e4m3(a) | (f32_to_e4m3(b) << 8) |
           (f32_to_e4m3(c) << 16) | (f32_to_e4m3(d) << 24);
}

__global__ void quant_rowmajor_kernel(const float* __restrict__ src,
                                      unsigned int* __restrict__ dst, int n4) {
    int i = blockIdx.x * blockDim.x + threadIdx.x;
    if (i < n4) {
        float4 f = ((const float4*)src)[i];
        dst[i] = pack4_e4m3(f.x, f.y, f.z, f.w);
    }
}

__global__ void quant_transpose_kernel(const float* __restrict__ B,
                                       unsigned char* __restrict__ BT) {
    __shared__ float tile[32][33];
    const int n0 = blockIdx.x * 32;
    const int k0 = blockIdx.y * 32;
    const int tx = threadIdx.x;
    const int ty = threadIdx.y;
    #pragma unroll
    for (int i = 0; i < 32; i += 8)
        tile[ty + i][tx] = B[(size_t)(k0 + ty + i) * N_DIM + n0 + tx];
    __syncthreads();
    const int lin = ty * 32 + tx;
    const int kq  = lin & 7;
    const int nr  = lin >> 3;
    unsigned int p = pack4_e4m3(tile[kq * 4 + 0][nr], tile[kq * 4 + 1][nr],
                                tile[kq * 4 + 2][nr], tile[kq * 4 + 3][nr]);
    *(unsigned int*)(BT + (size_t)(n0 + nr) * K_DIM + k0 + kq * 4) = p;
}

__global__ void __launch_bounds__(256)
fp8_gemm_kernel(const unsigned char* __restrict__ Aq,
                const unsigned char* __restrict__ BTq,
                __hip_bfloat16* __restrict__ C) {
    __shared__ unsigned char Abuf[2][BLK_M * PA];
    __shared__ unsigned char Bbuf[2][BLK_N * PB];

    const int tid  = threadIdx.x;
    const int lane = tid & 31;
    const int wave = tid >> 5;
    const int lr   = lane & 15;
    const int half = lane >> 4;

    const int wm = wave >> 2;
    const int wn = wave & 3;
    const int mBlock = blockIdx.y * BLK_M;
    const int nBlock = blockIdx.x * BLK_N;

    auto stage = [&](int buf, int kk) {
        #pragma unroll
        for (int i = 0; i < 4; ++i) {
            const int chunk = tid + i * 256;
            const int row   = chunk >> 3;
            const int cir   = chunk & 7;
            ASYNC_CP16(&Abuf[buf][row * PA + cir * 16],
                       Aq  + (size_t)(mBlock + row) * K_DIM + kk + cir * 16);
            ASYNC_CP16(&Bbuf[buf][row * PB + cir * 16],
                       BTq + (size_t)(nBlock + row) * K_DIM + kk + cir * 16);
        }
    };

    v8f acc[4][2];
    #pragma unroll
    for (int mt = 0; mt < 4; ++mt)
        #pragma unroll
        for (int nt = 0; nt < 2; ++nt)
            acc[mt][nt] = (v8f){};

    union Frag { v16i v; uint2 d[8]; uint4 q[4]; };

    stage(0, 0);
    WAIT_ASYNC();
    __syncthreads();

    for (int ks = 0; ks < K_DIM / BLK_K; ++ks) {
        const int cur = ks & 1;
        if (ks + 1 < K_DIM / BLK_K)
            stage(cur ^ 1, (ks + 1) * BLK_K);

        const unsigned char* Ab = &Abuf[cur][0];
        const unsigned char* Bb = &Bbuf[cur][0];

        Frag b[2];
        #pragma unroll
        for (int nt = 0; nt < 2; ++nt) {
            const unsigned char* bp = Bb + (wn * 32 + nt * 16 + lr) * PB + half * 16;
            #pragma unroll
            for (int j = 0; j < 4; ++j)
                b[nt].q[j] = *(const uint4*)(bp + 32 * j);
        }
        #pragma unroll
        for (int mt = 0; mt < 4; ++mt) {
            Frag a;
            const unsigned char* ap = Ab + (wm * 64 + mt * 16 + lr) * PA + half * 16;
            #pragma unroll
            for (int j = 0; j < 8; ++j)
                a.d[j] = *(const uint2*)(ap + 32 * (j >> 1) + 8 * (j & 1));
            acc[mt][0] = __builtin_amdgcn_wmma_f32_16x16x128_fp8_fp8(
                             a.v, b[0].v, (short)0, acc[mt][0], false, false);
            acc[mt][1] = __builtin_amdgcn_wmma_f32_16x16x128_fp8_fp8(
                             a.v, b[1].v, (short)0, acc[mt][1], false, false);
        }

        WAIT_ASYNC();
        __syncthreads();
    }

    #pragma unroll
    for (int mt = 0; mt < 4; ++mt)
        #pragma unroll
        for (int nt = 0; nt < 2; ++nt) {
            const int col = nBlock + wn * 32 + nt * 16 + lr;
            #pragma unroll
            for (int r = 0; r < 8; ++r) {
                const int row = mBlock + wm * 64 + mt * 16 + half * 8 + r;
                C[(size_t)row * N_DIM + col] = __float2bfloat16(acc[mt][nt][r]);
            }
        }
}

extern "C" void kernel_launch(void* const* d_in, const int* in_sizes, int n_in,
                              void* d_out, int out_size, void* d_ws, size_t ws_size,
                              hipStream_t stream) {
    const float* A = (const float*)d_in[0];
    const float* B = (const float*)d_in[1];
    __hip_bfloat16* C = (__hip_bfloat16*)d_out;

    unsigned char* Aq  = (unsigned char*)d_ws;
    unsigned char* BTq = Aq + (size_t)M_DIM * K_DIM;

    const int n4A = (M_DIM * K_DIM) / 4;
    quant_rowmajor_kernel<<<n4A / 256, 256, 0, stream>>>(A, (unsigned int*)Aq, n4A);

    dim3 tb(32, 8);
    dim3 gb(N_DIM / 32, K_DIM / 32);
    quant_transpose_kernel<<<gb, tb, 0, stream>>>(B, BTq);

    dim3 gg(N_DIM / BLK_N, M_DIM / BLK_M);
    fp8_gemm_kernel<<<gg, 256, 0, stream>>>(Aq, BTq, C);
}
